// KNN_65369402245845
// MI455X (gfx1250) — matrix operand order repaired and checked by reading, not compiled, not run on hardware
//
#include <hip/hip_runtime.h>
#include <hip/hip_bf16.h>

// ---------------- problem constants (fixed by reference setup_inputs) -------
#define DD        128        // feature dim
#define NQ        1024       // queries
#define ND        50000      // database points
#define KNN_K     16
#define NCLS      10
#define NS        125        // N-slices (50000/16 = 3125 tiles = 25 * 125)
#define TILES_PER_SLICE 25
#define CAND_TOTAL (NS * KNN_K)   // 2000 partial candidates per query
#define QPB       128        // queries per block (4 waves x 32)

// ---------------- workspace layout (bytes) ----------------------------------
#define OFF_DH  ((size_t)0)                    // data hi bf16  : 50000*128*2
#define OFF_DL  ((size_t)12800000)             // data lo bf16
#define OFF_DD  ((size_t)25600000)             // data norms f32: 50000*4
#define OFF_XH  ((size_t)25800192)             // x hi bf16     : 1024*128*2
#define OFF_XL  ((size_t)26062336)             // x lo bf16
#define OFF_XX  ((size_t)26324480)             // x norms f32
#define OFF_PD  ((size_t)26328576)             // partial dists : 1024*2000*4
#define OFF_PI  ((size_t)34520576)             // partial idx   : 1024*2000*4
// total ~42.7 MB

typedef __attribute__((ext_vector_type(16))) __bf16 v16bf;
typedef __attribute__((ext_vector_type(8)))  __bf16 v8bf;
typedef __attribute__((ext_vector_type(8)))  float  v8f;
typedef __attribute__((ext_vector_type(4)))  unsigned v4u;
typedef __attribute__((ext_vector_type(8)))  int      v8i;
typedef __attribute__((ext_vector_type(4)))  int      v4i;

union Frag { v16bf v; v8bf h[2]; };

__device__ __forceinline__ unsigned short f32_to_bf16_rne(float f) {
  unsigned u = __float_as_uint(f);
  unsigned r = u + 0x7FFFu + ((u >> 16) & 1u);
  return (unsigned short)(r >> 16);
}
__device__ __forceinline__ float bf16u_to_f32(unsigned short h) {
  return __uint_as_float(((unsigned)h) << 16);
}

// LDS byte offset of a __shared__ object: flat address low 32 bits are the
// wave-relative LDS address (ISA 10.2 aperture mapping).
__device__ __forceinline__ unsigned lds_off(const void* p) {
  return (unsigned)(unsigned long long)(uintptr_t)p;
}

// ---- TDM: 1-D contiguous tile (nelem bf16 elements) global -> LDS ----------
// D# per ISA 8.3/8.4: group0 = {flags(count=1), lds_addr, global_addr, type=2},
// group1 = {data_size=2B, tensor_dim0=nelem, tensor_dim1=1, tile_dim0=nelem,
//           tile_dim1/2=0(unused), stride0=nelem}. Groups 2/3 zero (<=2-D).
__device__ __forceinline__ void tdm_load_1d(const void* gptr, unsigned laddr,
                                            unsigned nelem) {
  unsigned long long ga = (unsigned long long)(uintptr_t)gptr;
  v4u g0;
  g0[0] = 1u;                                             // count=1, user desc
  g0[1] = laddr;                                          // lds_addr
  g0[2] = (unsigned)(ga & 0xFFFFFFFFu);                   // global_addr lo
  g0[3] = (unsigned)((ga >> 32) & 0x01FFFFFFu) | (2u << 30); // ga hi | type=2
  v8i g1;
  g1[0] = (int)(1u << 16);                                // data_size=1 (2B)
  g1[1] = (int)((nelem & 0xFFFFu) << 16);                 // tensor_dim0 lo16
  g1[2] = (int)((nelem >> 16) & 0xFFFFu) | (1 << 16);     // td0 hi16 | td1=1
  g1[3] = (int)((nelem & 0xFFFFu) << 16);                 // tile_dim0 = nelem
  g1[4] = 0;                                              // tile_dim1/2 unused
  g1[5] = (int)nelem;                                     // dim0_stride lo32
  g1[6] = 0;
  g1[7] = 0;
  v4i gz = (v4i)0;
#if __clang_major__ >= 23
  v8i gz8 = (v8i)0;
  __builtin_amdgcn_tensor_load_to_lds(g0, g1, gz, gz, gz8, 0);
#else
  __builtin_amdgcn_tensor_load_to_lds(g0, g1, gz, gz, 0);
#endif
}

// ---------------- prep: f32 -> (hi,lo) bf16 + row squared norms --------------
__global__ __launch_bounds__(128)
void knn_prep(const float* __restrict__ src,
              unsigned short* __restrict__ oh, unsigned short* __restrict__ ol,
              float* __restrict__ norms)
{
  __shared__ float red[4];
  const int row = blockIdx.x;
  const int t = threadIdx.x;
  float v = src[(size_t)row * DD + t];
  unsigned short h = f32_to_bf16_rne(v);
  float hv = bf16u_to_f32(h);
  unsigned short l = f32_to_bf16_rne(v - hv);   // residual, exact in f32
  oh[(size_t)row * DD + t] = h;
  ol[(size_t)row * DD + t] = l;
  float s = v * v;
  #pragma unroll
  for (int o = 16; o; o >>= 1) s += __shfl_xor(s, o, 32);
  if ((t & 31) == 0) red[t >> 5] = s;
  __syncthreads();
  if (t == 0) norms[row] = red[0] + red[1] + red[2] + red[3];
}

// ------- fused split-bf16 WMMA GEMM + per-slice exact top-16 ----------------
// 4 waves / block share a TDM-staged, double-buffered B tile in LDS.
__global__ __launch_bounds__(QPB)
void knn_gemm_topk(const __bf16* __restrict__ dh, const __bf16* __restrict__ dl,
                   const float* __restrict__ dd,
                   const __bf16* __restrict__ xh, const __bf16* __restrict__ xl,
                   const float* __restrict__ xx,
                   float* __restrict__ part_d, int* __restrict__ part_i)
{
  __shared__ __align__(128) __bf16 bufh[2][16 * DD];   // 2 x 4KB, data-hi tile
  __shared__ __align__(128) __bf16 bufl[2][16 * DD];   // 2 x 4KB, data-lo tile
  __shared__ float lds_d2[4][32 * 16];
  __shared__ float topd[4][32 * 16];
  __shared__ int   topi[4][32 * 16];

  const int lane = threadIdx.x & 31;
  const int wv   = threadIdx.x >> 5;          // wave id 0..3
  const int slice = blockIdx.y;               // 0..124: 25 tiles each
  const int q0   = blockIdx.x * QPB + wv * 32;
  const int col  = lane & 15;
  const int hf   = lane >> 4;

  // ---- A fragments (16x32 bf16, ISA layout): resident for whole slice ----
  Frag a0h[4], a0l[4], a1h[4], a1l[4];
  const int offA = hf * 8;
  #pragma unroll
  for (int f = 0; f < 4; ++f) {
    const __bf16* p;
    p = xh + (size_t)(q0 + col) * DD + f * 32 + offA;
    a0h[f].h[0] = *(const v8bf*)p;        a0h[f].h[1] = *(const v8bf*)(p + 16);
    p = xl + (size_t)(q0 + col) * DD + f * 32 + offA;
    a0l[f].h[0] = *(const v8bf*)p;        a0l[f].h[1] = *(const v8bf*)(p + 16);
    p = xh + (size_t)(q0 + 16 + col) * DD + f * 32 + offA;
    a1h[f].h[0] = *(const v8bf*)p;        a1h[f].h[1] = *(const v8bf*)(p + 16);
    p = xl + (size_t)(q0 + 16 + col) * DD + f * 32 + offA;
    a1l[f].h[0] = *(const v8bf*)p;        a1l[f].h[1] = *(const v8bf*)(p + 16);
  }

  // per-lane x-norms matching the C/D layout: reg r, lane -> M = r + 8*hf
  float xxv0[8], xxv1[8];
  #pragma unroll
  for (int r = 0; r < 8; ++r) {
    xxv0[r] = xx[q0 + r + 8 * hf];
    xxv1[r] = xx[q0 + 16 + r + 8 * hf];
  }

  // init per-query top-16 (lane owns query q0+lane within its wave's region)
  #pragma unroll
  for (int s = 0; s < 16; ++s) {
    topd[wv][lane * 16 + s] = 3.0e38f;
    topi[wv][lane * 16 + s] = 0;
  }
  float curmax = 3.0e38f; int maxslot = 0;

  const int tile0 = slice * TILES_PER_SLICE;

  // prologue: TDM-stage tile 0 into buffer 0 (wave 0 drives the TDM)
  if (wv == 0) {
    tdm_load_1d(dh + (size_t)(tile0 * 16) * DD, lds_off(&bufh[0][0]), 16 * DD);
    tdm_load_1d(dl + (size_t)(tile0 * 16) * DD, lds_off(&bufl[0][0]), 16 * DD);
  }

  for (int t = 0; t < TILES_PER_SLICE; ++t) {
    const int cur = t & 1;
    const int nb  = (tile0 + t) * 16;

    if (wv == 0) __builtin_amdgcn_s_wait_tensorcnt(0);  // buf[cur] landed
    __syncthreads();                                    // visible to all waves
    if (wv == 0 && t + 1 < TILES_PER_SLICE) {           // overlap next copy
      const int nn = (tile0 + t + 1) * 16;
      tdm_load_1d(dh + (size_t)nn * DD, lds_off(&bufh[cur ^ 1][0]), 16 * DD);
      tdm_load_1d(dl + (size_t)nn * DD, lds_off(&bufl[cur ^ 1][0]), 16 * DD);
    }

    // ---- B fragments from LDS (two 16-byte chunks of one data row: k 8hf.., 16+8hf..) -------
    const __bf16* ph = &bufh[cur][col * DD + hf * 8];
    const __bf16* pl = &bufl[cur][col * DD + hf * 8];
    v16bf bh[4], bl[4];
    #pragma unroll
    for (int f = 0; f < 4; ++f) {
      bh[f] = __builtin_shufflevector(*(const v8bf*)(ph + f * 32), *(const v8bf*)(ph + f * 32 + 16), 0,1,2,3,4,5,6,7,8,9,10,11,12,13,14,15);
      bl[f] = __builtin_shufflevector(*(const v8bf*)(pl + f * 32), *(const v8bf*)(pl + f * 32 + 16), 0,1,2,3,4,5,6,7,8,9,10,11,12,13,14,15);
    }
    const float ddv = dd[nb + col];

    // ---- split-bf16 product: x.d = xh.dh + xl.dh + xh.dl (f32 accumulate)
    v8f acc0 = {}; v8f acc1 = {};
    #pragma unroll
    for (int f = 0; f < 4; ++f) {
      acc0 = __builtin_amdgcn_wmma_f32_16x16x32_bf16(false, a0h[f].v, false, bh[f], (short)0, acc0, false, false);
      acc1 = __builtin_amdgcn_wmma_f32_16x16x32_bf16(false, a1h[f].v, false, bh[f], (short)0, acc1, false, false);
      acc0 = __builtin_amdgcn_wmma_f32_16x16x32_bf16(false, a0l[f].v, false, bh[f], (short)0, acc0, false, false);
      acc1 = __builtin_amdgcn_wmma_f32_16x16x32_bf16(false, a1l[f].v, false, bh[f], (short)0, acc1, false, false);
      acc0 = __builtin_amdgcn_wmma_f32_16x16x32_bf16(false, a0h[f].v, false, bl[f], (short)0, acc0, false, false);
      acc1 = __builtin_amdgcn_wmma_f32_16x16x32_bf16(false, a1h[f].v, false, bl[f], (short)0, acc1, false, false);
    }

    // ---- d2 = |x|^2 + |d|^2 - 2 x.d, scatter to LDS per C/D layout --------
    #pragma unroll
    for (int r = 0; r < 8; ++r) {
      float d0 = xxv0[r] + ddv - 2.0f * acc0[r]; d0 = d0 < 0.0f ? 0.0f : d0;
      float d1 = xxv1[r] + ddv - 2.0f * acc1[r]; d1 = d1 < 0.0f ? 0.0f : d1;
      lds_d2[wv][(r + 8 * hf) * 16 + col]      = d0;
      lds_d2[wv][(16 + r + 8 * hf) * 16 + col] = d1;
    }
    __syncthreads();   // also orders per-wave d2 store -> read

    // ---- selection: lane scans its query's 16 candidates ------------------
    for (int j = 0; j < 16; ++j) {
      float d = lds_d2[wv][lane * 16 + j];
      if (d < curmax) {                      // rare after warm-up
        topd[wv][lane * 16 + maxslot] = d;
        topi[wv][lane * 16 + maxslot] = nb + j;
        float mx = -1.0f; int ms = 0;
        #pragma unroll
        for (int s = 0; s < 16; ++s) {
          float v = topd[wv][lane * 16 + s];
          if (v > mx) { mx = v; ms = s; }
        }
        curmax = mx; maxslot = ms;
      }
    }
    // per-wave lds_d2 reuse next tile is ordered by in-order DS within a wave;
    // cross-wave buffers are ordered by the barrier at loop top.
  }

  // ---- write this slice's top-16 candidates ------------------------------
  const int q = q0 + lane;
  #pragma unroll
  for (int s = 0; s < 16; ++s) {
    part_d[(size_t)q * CAND_TOTAL + slice * 16 + s] = topd[wv][lane * 16 + s];
    part_i[(size_t)q * CAND_TOTAL + slice * 16 + s] = topi[wv][lane * 16 + s];
  }
}

// ---------------- merge partials, vote, argmax -------------------------------
__global__ __launch_bounds__(32)
void knn_merge_vote(const float* __restrict__ part_d, const int* __restrict__ part_i,
                    const int* __restrict__ label, float* __restrict__ out)
{
  __shared__ float cd[CAND_TOTAL];
  __shared__ int   ci[CAND_TOTAL];
  __shared__ int   win[KNN_K];
  const int q = blockIdx.x;
  const int lane = threadIdx.x;

  for (int j = lane; j < CAND_TOTAL; j += 32) {
    cd[j] = part_d[(size_t)q * CAND_TOTAL + j];
    ci[j] = part_i[(size_t)q * CAND_TOTAL + j];
  }
  __syncthreads();

  // 16 rounds of wave-wide argmin on lexicographic key (dist_bits, idx):
  // matches jax.lax.top_k tie-breaking (lower index first).
  for (int t = 0; t < KNN_K; ++t) {
    unsigned long long best = ~0ULL; int bslot = -1;
    for (int j = lane; j < CAND_TOTAL; j += 32) {
      unsigned long long k =
          (((unsigned long long)__float_as_uint(cd[j])) << 32) | (unsigned)ci[j];
      if (k < best) { best = k; bslot = j; }
    }
    #pragma unroll
    for (int o = 16; o; o >>= 1) {
      unsigned ohi = __shfl_xor((unsigned)(best >> 32), o, 32);
      unsigned olo = __shfl_xor((unsigned)best, o, 32);
      int os = __shfl_xor(bslot, o, 32);
      unsigned long long ob = (((unsigned long long)ohi) << 32) | olo;
      if (ob < best) { best = ob; bslot = os; }
    }
    if (lane == 0) { win[t] = (int)(best & 0xFFFFFFFFULL); cd[bslot] = 3.0e38f; }
    __syncthreads();
  }

  const int lab = (lane < KNN_K) ? label[win[lane]] : -1;
  int bestc = 0, bestn = -1;
  #pragma unroll
  for (int c = 0; c < NCLS; ++c) {
    int n = (int)__popcll(__ballot(lab == c));
    if (n > bestn) { bestn = n; bestc = c; }   // first max wins, like argmax
  }
  if (lane == 0) out[q] = (float)bestc;        // MIN_INDEX == 0
}

// ---------------- launch ------------------------------------------------------
extern "C" void kernel_launch(void* const* d_in, const int* in_sizes, int n_in,
                              void* d_out, int out_size, void* d_ws, size_t ws_size,
                              hipStream_t stream) {
  const float* x    = (const float*)d_in[0];   // [1024,128] f32
  const float* data = (const float*)d_in[1];   // [50000,128] f32
  const int*   lab  = (const int*)d_in[2];     // [50000] i32

  char* ws = (char*)d_ws;
  unsigned short* dh = (unsigned short*)(ws + OFF_DH);
  unsigned short* dl = (unsigned short*)(ws + OFF_DL);
  float*          dd = (float*)(ws + OFF_DD);
  unsigned short* xh = (unsigned short*)(ws + OFF_XH);
  unsigned short* xl = (unsigned short*)(ws + OFF_XL);
  float*          xx = (float*)(ws + OFF_XX);
  float*          pd = (float*)(ws + OFF_PD);
  int*            pi = (int*)(ws + OFF_PI);

  knn_prep<<<ND, 128, 0, stream>>>(data, dh, dl, dd);
  knn_prep<<<NQ, 128, 0, stream>>>(x, xh, xl, xx);

  knn_gemm_topk<<<dim3(NQ / QPB, NS), QPB, 0, stream>>>(
      (const __bf16*)dh, (const __bf16*)dl, dd,
      (const __bf16*)xh, (const __bf16*)xl, xx, pd, pi);

  knn_merge_vote<<<NQ, 32, 0, stream>>>(pd, pi, lab, (float*)d_out);
}
